// AttentionBasedPooling_55525337202919
// MI455X (gfx1250) — matrix operand order repaired and checked by reading, not compiled, not run on hardware
//
#include <hip/hip_runtime.h>

typedef __attribute__((ext_vector_type(16))) _Float16 v16h;
typedef __attribute__((ext_vector_type(8)))  float    v8f;

namespace {
constexpr int kF = 32;           // fields
constexpr int kD = 64;           // embed dim (GEMM K)
constexpr int kH = 64;           // hidden dim (GEMM N)
constexpr int kP = 496;          // kF*(kF-1)/2 pairs (GEMM M)
constexpr int kMTiles = 31;      // kP / 16, exact
constexpr int kXStride = 68;     // f32 row stride (pad: 68 % 64 banks != 0)
constexpr int kWStride = 66;     // f16 row stride (pad)
constexpr int kThreads = 256;    // 8 waves (wave32)
constexpr int kWaves = 8;
}

__global__ __launch_bounds__(kThreads)
void afm_attn_pool_kernel(const float* __restrict__ x,
                          const float* __restrict__ W1,
                          const float* __restrict__ b1,
                          const float* __restrict__ Ws,
                          float* __restrict__ out)
{
    __shared__ float     xs[kF * kXStride];     // x[b] f32, padded rows
    __shared__ _Float16  W1h[kD * kWStride];    // W1 as f16 [d][h], padded rows
    __shared__ float     scores[kP];
    __shared__ float     dots[kP];              // dot(x_i, x_j) per pair
    __shared__ float     b1s[kH], wss[kH];
    __shared__ unsigned char pi[kP], pj[kP];
    __shared__ float     redA[kWaves], redB[kWaves], redC[kWaves];

    const int tid  = threadIdx.x;
    const int lane = tid & 31;
    const int ln   = lane & 15;   // column / row-within-half index
    const int hk   = lane >> 4;   // lane-half selector
    const int wid  = tid >> 5;    // wave id
    const int b    = blockIdx.x;

    // ---- stage x[b] into LDS (float4 loads, padded f32 rows) ----
    const float4* xg = reinterpret_cast<const float4*>(x + (size_t)b * (kF * kD));
    for (int v = tid; v < kF * kD / 4; v += kThreads) {
        int f = v >> 4, c = v & 15;                 // 16 float4 per 64-elem row
        *reinterpret_cast<float4*>(&xs[f * kXStride + 4 * c]) = xg[v];
    }
    // ---- W1 -> f16 LDS ----
    for (int v = tid; v < kD * kH; v += kThreads) {
        int k = v >> 6, n = v & 63;
        W1h[k * kWStride + n] = (_Float16)W1[v];
    }
    if (tid < kH) { b1s[tid] = b1[tid]; wss[tid] = Ws[tid]; }
    // ---- triu(k=1) pair table ----
    for (int p = tid; p < kP; p += kThreads) {
        int i = 0, rem = p;
        while (rem >= (kF - 1 - i)) { rem -= (kF - 1 - i); ++i; }
        pi[p] = (unsigned char)i;
        pj[p] = (unsigned char)(i + 1 + rem);
    }
    __syncthreads();

    // ---- dots[p] = dot(x_i, x_j)  (== row-sum of cross[p,:], used for pooling) ----
    for (int p = tid; p < kP; p += kThreads) {
        const float4* xi = reinterpret_cast<const float4*>(&xs[pi[p] * kXStride]);
        const float4* xj = reinterpret_cast<const float4*>(&xs[pj[p] * kXStride]);
        float acc = 0.f;
        #pragma unroll
        for (int q = 0; q < kD / 4; ++q) {
            float4 a = xi[q], c = xj[q];
            acc += a.x * c.x + a.y * c.y + a.z * c.z + a.w * c.w;
        }
        dots[p] = acc;
    }

    // ---- preload B fragments: W1 tiles for all 4 N-tiles x 2 K-steps ----
    // 16-bit B 32x16 layout (matches A): VGPR v<4, lanes 0-15 hold K=2v,2v+1; lanes 16-31 K=8+2v,8+2v+1; v>=4 adds 8 more
    v16h bf[4][2];
    float b1v[4], wsv[4];
    #pragma unroll
    for (int nt = 0; nt < 4; ++nt) {
        b1v[nt] = b1s[nt * 16 + ln];
        wsv[nt] = wss[nt * 16 + ln];
        #pragma unroll
        for (int ks = 0; ks < 2; ++ks) {
            #pragma unroll
            for (int v = 0; v < 8; ++v) {
                int kb = (v < 4 ? 2 * v : 2 * v + 8) + 8 * hk + 32 * ks;
                bf[nt][ks][2 * v]     = W1h[kb * kWStride + nt * 16 + ln];
                bf[nt][ks][2 * v + 1] = W1h[(kb + 1) * kWStride + nt * 16 + ln];
            }
        }
    }

    // 16-bit A 16x32 layout: K-pair bases per VGPR, lane-half adds 8
    const int klist[8] = {0, 2, 4, 6, 16, 18, 20, 22};

    // ---- GEMM h = relu(cross @ W1 + b1), fused scores = h @ Ws ----
    // 31 M-tiles, wave-uniform loop (EXEC all ones around WMMA)
    for (int tile = wid; tile < kMTiles; tile += kWaves) {
        int p = tile * 16 + ln;                      // A rows M=0..15 = ln for both halves
        const float* xi = &xs[pi[p] * kXStride];
        const float* xj = &xs[pj[p] * kXStride];
        v16h a0, a1;
        #pragma unroll
        for (int v = 0; v < 8; ++v) {
            int kb = klist[v] + 8 * hk;
            a0[2 * v]     = (_Float16)(xi[kb]      * xj[kb]);
            a0[2 * v + 1] = (_Float16)(xi[kb + 1]  * xj[kb + 1]);
            a1[2 * v]     = (_Float16)(xi[kb + 32] * xj[kb + 32]);
            a1[2 * v + 1] = (_Float16)(xi[kb + 33] * xj[kb + 33]);
        }
        float sacc[8] = {0, 0, 0, 0, 0, 0, 0, 0};
        #pragma unroll
        for (int nt = 0; nt < 4; ++nt) {
            v8f acc = {};
            acc = __builtin_amdgcn_wmma_f32_16x16x32_f16(false, a0, false, bf[nt][0],
                                                         (short)0, acc, false, false);
            acc = __builtin_amdgcn_wmma_f32_16x16x32_f16(false, a1, false, bf[nt][1],
                                                         (short)0, acc, false, false);
            #pragma unroll
            for (int r = 0; r < 8; ++r)
                sacc[r] += fmaxf(acc[r] + b1v[nt], 0.f) * wsv[nt];
        }
        // reduce over N (16 lanes of each half hold columns 0..15)
        #pragma unroll
        for (int mask = 1; mask <= 8; mask <<= 1)
            #pragma unroll
            for (int r = 0; r < 8; ++r)
                sacc[r] += __shfl_xor(sacc[r], mask, 32);
        if (ln == 0) {
            #pragma unroll
            for (int r = 0; r < 8; ++r)
                scores[tile * 16 + 8 * hk + r] = sacc[r];   // D: vgpr r -> M = r + 8*hk
        }
    }
    __syncthreads();

    // ---- softmax over pairs + weighted sum of dots (bs dropped: softmax-invariant) ----
    float m = -3.0e38f;
    for (int p = tid; p < kP; p += kThreads) m = fmaxf(m, scores[p]);
    #pragma unroll
    for (int mask = 16; mask >= 1; mask >>= 1) m = fmaxf(m, __shfl_xor(m, mask, 32));
    if (lane == 0) redA[wid] = m;
    __syncthreads();
    float mg = redA[0];
    #pragma unroll
    for (int w = 1; w < kWaves; ++w) mg = fmaxf(mg, redA[w]);

    float se = 0.f, sed = 0.f;
    for (int p = tid; p < kP; p += kThreads) {
        float e = __expf(scores[p] - mg);
        se  += e;
        sed += e * dots[p];
    }
    #pragma unroll
    for (int mask = 16; mask >= 1; mask >>= 1) {
        se  += __shfl_xor(se,  mask, 32);
        sed += __shfl_xor(sed, mask, 32);
    }
    if (lane == 0) { redB[wid] = se; redC[wid] = sed; }
    __syncthreads();
    if (tid == 0) {
        float st = 0.f, sdt = 0.f;
        #pragma unroll
        for (int w = 0; w < kWaves; ++w) { st += redB[w]; sdt += redC[w]; }
        out[b] = sdt / st;
    }
}

extern "C" void kernel_launch(void* const* d_in, const int* in_sizes, int n_in,
                              void* d_out, int out_size, void* d_ws, size_t ws_size,
                              hipStream_t stream) {
    const float* x  = (const float*)d_in[0];
    const float* W1 = (const float*)d_in[1];
    const float* b1 = (const float*)d_in[2];
    const float* Ws = (const float*)d_in[3];
    // d_in[4] (bs) shifts every score equally -> cancels in softmax; ignored.
    const int B = in_sizes[0] / (kF * kD);
    afm_attn_pool_kernel<<<B, kThreads, 0, stream>>>(x, W1, b1, Ws, (float*)d_out);
}
